// PolyAdapter_28527172780240
// MI455X (gfx1250) — matrix operand order repaired and checked by reading, not compiled, not run on hardware
//
#include <hip/hip_runtime.h>

// ---------------------------------------------------------------------------
// PolyAdapter fused kernel for gfx1250 (MI455X), wave32 + WMMA.
//   out = ia3 * (x@Wb^T + bb + 4*(x@A)@B) + sum_e w[token,e]*(x@We[e]^T + be[e])
// with w = top-1 softmax router weights (zeros elsewhere -> dense-weighted
// expert combine is numerically identical to the reference).
// Round 4: avoid LDS-pointer static initializer (buffer select via offset).
// ---------------------------------------------------------------------------

typedef __attribute__((ext_vector_type(16))) __bf16        v16bf;
typedef __attribute__((ext_vector_type(8)))  float         v8f;
typedef __attribute__((ext_vector_type(2)))  float         v2f;
typedef __attribute__((ext_vector_type(4)))  unsigned int  uint4v;
typedef __attribute__((ext_vector_type(4)))  int           int4v;
typedef __attribute__((ext_vector_type(4)))  float         float4v;
typedef __attribute__((ext_vector_type(4)))  unsigned short ushort4v;

#define IN_F   512
#define OUT_F  512
#define NEXP   4
#define TM     128          // tokens per workgroup
#define XROW   520          // padded bf16 row stride (bank-conflict padding, 16B aligned)
#define NMAT   5            // Wb + 4 experts
#define ENT1   (NMAT*8*4*16*32)      // 81920 lane-entries, 32B each (GEMM weights)
#define ENT2   (16*32)               // router+LoRA combined matrix lane-entries
#define SM_TSC   133120              // byte offset of tsc   (128*520*2)
#define SM_WEX   141312              // byte offset of wex   (+128*16*4)
#define SM_BST0  143360              // byte offset of stage buffer 0 (+128*4*4)
#define SM_TOTAL (SM_BST0 + 2*65536) // 274432 bytes dynamic LDS (< 320KB WGP)
#define NSTAGE   (8 * NMAT)          // 40 weight stages (nc-major, mat-minor)

#if __has_builtin(__builtin_amdgcn_global_load_async_to_lds_b128)
#define HAVE_ASYNC_LDS 1
#else
#define HAVE_ASYNC_LDS 0
#endif

typedef __attribute__((address_space(1))) int4v* gv4p;   // global v4i*
typedef __attribute__((address_space(3))) int4v* lv4p;   // LDS v4i*

__device__ __forceinline__ unsigned short f2bf(float f) {
    unsigned int u = __float_as_uint(f);
    return (unsigned short)((u + 0x7FFFu + ((u >> 16) & 1u)) >> 16);  // RNE
}

__device__ __forceinline__ void wait_async_mine() {
#if HAVE_ASYNC_LDS
#if __has_builtin(__builtin_amdgcn_s_wait_asynccnt)
    __builtin_amdgcn_s_wait_asynccnt(0);
#else
    asm volatile("s_wait_asynccnt 0x0" ::: "memory");
#endif
#endif
}

// Issue one 64KB weight stage: global (L2-resident bf16 pack) -> LDS buffer.
// Async path: 16 x global_load_async_to_lds_b128 per thread, no VGPR data,
// tracked by ASYNCcnt. Fallback: plain load/ds_store (ordered by DScnt).
__device__ __forceinline__ void stage_issue(const unsigned short* __restrict__ wpk,
                                            int blockIdxW, char* dstbuf, int tid) {
    char* src = const_cast<char*>((const char*)wpk) + (size_t)blockIdxW * 65536;
#if HAVE_ASYNC_LDS
    #pragma unroll
    for (int it = 0; it < 16; ++it) {
        int c = (tid + it * 256) * 16;
        __builtin_amdgcn_global_load_async_to_lds_b128(
            (gv4p)(src + c), (lv4p)(dstbuf + c), 0, 0);
    }
#else
    const uint4v* s4 = (const uint4v*)src;
    uint4v* d4 = (uint4v*)dstbuf;
    #pragma unroll 4
    for (int it = 0; it < 16; ++it)
        d4[tid + it * 256] = s4[tid + it * 256];
#endif
}

// ---------------------------------------------------------------------------
// Kernel 0: repack weights to bf16 in CDNA5 WMMA B-matrix lane layout.
// B (32x16, 16-bit): this pack gives lane l col n=l%16, K = (l/16)*16 .. +15 contiguous
// (same k order as A; not the ISA table order). Each lane-entry = 16 bf16 = 32 bytes, stored so one
// (matrix, 64-col chunk) = contiguous 64KB for fast LDS staging.
// ---------------------------------------------------------------------------
__global__ void repack_weights(const float* __restrict__ Wb,
                               const float* __restrict__ We,
                               const float* __restrict__ Wr,
                               const float* __restrict__ A,
                               unsigned short* __restrict__ ws) {
    int gid = blockIdx.x * blockDim.x + threadIdx.x;
    if (gid < ENT1) {
        int lane = gid & 31;
        int k    = (gid >> 5) & 15;   // K-tile (32 K each)
        int lt   = (gid >> 9) & 3;    // 16-col tile within chunk
        int nc   = (gid >> 11) & 7;   // 64-col chunk
        int mat  = (gid >> 14);       // 0=Wb, 1..4=experts
        const float* W = (mat == 0) ? Wb : (We + (size_t)(mat - 1) * OUT_F * IN_F);
        int o  = (nc * 4 + lt) * 16 + (lane & 15);
        int d0 = k * 32 + (lane >> 4) * 16;
        unsigned short tmp[16];
        #pragma unroll
        for (int i = 0; i < 16; ++i) tmp[i] = f2bf(W[(size_t)o * IN_F + d0 + i]);
        size_t entry = ((((size_t)(mat * 8 + nc) * 4 + lt) * 16 + k) * 32 + lane);
        uint4v* dst = (uint4v*)(ws + entry * 16);
        dst[0] = ((const uint4v*)tmp)[0];
        dst[1] = ((const uint4v*)tmp)[1];
    } else if (gid < ENT1 + ENT2) {
        // combined router (cols 0..3) + LoRA-A (cols 4..11) + zero pad matrix
        int e    = gid - ENT1;
        int lane = e & 31;
        int k    = e >> 5;
        int u    = lane & 15;
        int d0   = k * 32 + (lane >> 4) * 16;
        unsigned short tmp[16];
        #pragma unroll
        for (int i = 0; i < 16; ++i) {
            int d = d0 + i;
            float v = (u < 4) ? Wr[u * IN_F + d]
                    : (u < 12 ? A[d * 8 + (u - 4)] : 0.0f);
            tmp[i] = f2bf(v);
        }
        size_t entry = (size_t)ENT1 + e;
        uint4v* dst = (uint4v*)(ws + entry * 16);
        dst[0] = ((const uint4v*)tmp)[0];
        dst[1] = ((const uint4v*)tmp)[1];
    }
}

// A-fragment (16x32 bf16) from LDS: lane l -> row M=l%16, K = (l/16)*16 .. +15
// contiguous (matches B pack); two ds_load_b128.
__device__ __forceinline__ v16bf load_afrag(const unsigned short* xs, int rbase,
                                            int k, int lane) {
    int row = rbase + (lane & 15);
    int d0  = k * 32 + (lane >> 4) * 16;
    union { uint4v u[2]; v16bf v; } t;
    t.u[0] = *(const uint4v*)(xs + row * XROW + d0);
    t.u[1] = *(const uint4v*)(xs + row * XROW + d0 + 8);
    return t.v;
}

__global__ __launch_bounds__(256)
void poly_main(const float* __restrict__ x,
               const float* __restrict__ bb,
               const float* __restrict__ Blora,
               const float* __restrict__ ia3,
               const float* __restrict__ br,
               const float* __restrict__ be,
               const unsigned short* __restrict__ wpk,
               float* __restrict__ out) {
    extern __shared__ char smem[];
    unsigned short* xs  = (unsigned short*)smem;      // [128][520] bf16 x
    float* tsc          = (float*)(smem + SM_TSC);    // [128][16] logits(0..3)+t(4..11)
    float* wex          = (float*)(smem + SM_WEX);    // [128][4] top-1 weights
    // stage buffers: smem + SM_BST0 + (i&1)*65536 (selected by arithmetic,
    // NOT a pointer array -- avoids addrspacecast static initializers)

    const int tid   = threadIdx.x;
    const int lane  = tid & 31;
    const int wave  = tid >> 5;
    const int rbase = wave * 16;
    const int c16   = lane & 15;
    const int lh    = lane >> 4;
    const size_t blockM0 = (size_t)blockIdx.x * TM;

    // Prologue: kick off stage 0 (nc=0, mat=0 -> weight block index 0) ASAP
    // so the DMA overlaps the x conversion + router phases.
    stage_issue(wpk, 0, smem + SM_BST0, tid);

    // ---- Phase 1: load x block, convert f32 -> bf16 into LDS --------------
    {
        const float4v* xblk = (const float4v*)(x + blockM0 * IN_F);
        #pragma unroll 4
        for (int it = 0; it < (TM * IN_F / 4) / 256; ++it) {
            int idx4 = tid + it * 256;
            float4v v = xblk[idx4];
            int elem = idx4 * 4;
            int row  = elem >> 9;
            int c    = elem & (IN_F - 1);
            ushort4v h;
            h[0] = f2bf(v[0]); h[1] = f2bf(v[1]);
            h[2] = f2bf(v[2]); h[3] = f2bf(v[3]);
            *(ushort4v*)(xs + row * XROW + c) = h;
        }
    }
    __syncthreads();

    // ---- Phase 2: router logits + LoRA-t in ONE WMMA pass -----------------
    {
        v8f acc = {};
        for (int k = 0; k < 16; ++k) {
            v16bf a = load_afrag(xs, rbase, k, lane);
            union { uint4v u[2]; v16bf v; } t;
            const uint4v* p = (const uint4v*)(wpk + ((size_t)ENT1 + k * 32 + lane) * 16);
            t.u[0] = p[0]; t.u[1] = p[1];
            acc = __builtin_amdgcn_wmma_f32_16x16x32_bf16(false, a, false, t.v,
                                                          (short)0, acc, false, false);
        }
        float badd = (c16 < 4) ? br[c16] : 0.0f;
        #pragma unroll
        for (int j = 0; j < 8; ++j)
            tsc[(rbase + j + 8 * lh) * 16 + c16] = acc[j] + badd;
    }
    __syncthreads();

    // ---- Phase 3: per-token softmax + top-1 -------------------------------
    if (lane < 16) {
        int row = rbase + lane;
        float l0 = tsc[row * 16 + 0], l1 = tsc[row * 16 + 1];
        float l2 = tsc[row * 16 + 2], l3 = tsc[row * 16 + 3];
        float m  = fmaxf(fmaxf(l0, l1), fmaxf(l2, l3));
        float e0 = __expf(l0 - m), e1 = __expf(l1 - m);
        float e2 = __expf(l2 - m), e3 = __expf(l3 - m);
        float s  = e0 + e1 + e2 + e3;
        int arg = 0; float bm = e0;
        if (e1 > bm) { bm = e1; arg = 1; }
        if (e2 > bm) { bm = e2; arg = 2; }
        if (e3 > bm) { bm = e3; arg = 3; }
        float wv = bm / s;
        wex[row * 4 + 0] = (arg == 0) ? wv : 0.0f;
        wex[row * 4 + 1] = (arg == 1) ? wv : 0.0f;
        wex[row * 4 + 2] = (arg == 2) ? wv : 0.0f;
        wex[row * 4 + 3] = (arg == 3) ? wv : 0.0f;
    }
    __syncthreads();

    // per-row expert weights in D-matrix lane layout (row = j + 8*lh)
    float wrow[NEXP][8];
    #pragma unroll
    for (int e = 0; e < NEXP; ++e)
        #pragma unroll
        for (int j = 0; j < 8; ++j)
            wrow[e][j] = wex[(rbase + j + 8 * lh) * 4 + e];

    // ---- Phase 4: main GEMMs (Wb + 4 experts), 64 output cols at a time ---
    // Double-buffered stages: stage i = nc*5 + mat, weight block (mat*8 + nc).
    // Loop body: drain own async -> barrier publishes buf[i&1] AND proves all
    // waves finished reading buf[(i+1)&1] (consumed at stage i-1) -> issue
    // stage i+1 into it -> compute stage i while the DMA runs.
    for (int nc = 0; nc < 8; ++nc) {
        float baseacc[4][8], expacc[4][8];
        #pragma unroll
        for (int lt = 0; lt < 4; ++lt)
            #pragma unroll
            for (int j = 0; j < 8; ++j) { baseacc[lt][j] = 0.0f; expacc[lt][j] = 0.0f; }

        for (int mat = 0; mat < NMAT; ++mat) {
            const int i = nc * NMAT + mat;
            char* cur = smem + SM_BST0 + (size_t)(i & 1) * 65536;

            wait_async_mine();
            __syncthreads();

            if (i + 1 < NSTAGE) {
                int nn = (i + 1) / NMAT, nm = (i + 1) % NMAT;
                char* nxt = smem + SM_BST0 + (size_t)((i + 1) & 1) * 65536;
                stage_issue(wpk, nm * 8 + nn, nxt, tid);
            }

            #pragma unroll 1
            for (int lt = 0; lt < 4; ++lt) {
                v8f acc = {};
                for (int k = 0; k < 16; ++k) {
                    v16bf a = load_afrag(xs, rbase, k, lane);
                    union { uint4v u[2]; v16bf v; } t;
                    const uint4v* bp =
                        (const uint4v*)cur + (size_t)((lt * 16 + k) * 32 + lane) * 2;
                    t.u[0] = bp[0]; t.u[1] = bp[1];
                    acc = __builtin_amdgcn_wmma_f32_16x16x32_bf16(false, a, false, t.v,
                                                                  (short)0, acc, false, false);
                }
                if (mat == 0) {
                    #pragma unroll
                    for (int j = 0; j < 8; ++j) baseacc[lt][j] += acc[j];
                } else {
                    int e = mat - 1;
                    #pragma unroll
                    for (int j = 0; j < 8; ++j) expacc[lt][j] += wrow[e][j] * acc[j];
                }
            }
        }

        // ---- LoRA rank-8 tile in full-precision f32 WMMA + combine + store
        #pragma unroll 1
        for (int lt = 0; lt < 4; ++lt) {
            int ocol = nc * 64 + lt * 16 + c16;
            v8f lacc = {};
            #pragma unroll
            for (int kk = 0; kk < 2; ++kk) {
                int r0 = kk * 4 + lh * 2;              // K index within rank-8
                v2f af, bf2;
                af[0]  = tsc[(rbase + c16) * 16 + 4 + r0];
                af[1]  = tsc[(rbase + c16) * 16 + 4 + r0 + 1];
                bf2[0] = Blora[r0 * OUT_F + ocol];
                bf2[1] = Blora[(r0 + 1) * OUT_F + ocol];
                lacc = __builtin_amdgcn_wmma_f32_16x16x4_f32(false, af, false, bf2,
                                                             (short)0, lacc, false, false);
            }
            float ia  = ia3[ocol];
            float bbv = bb[ocol];
            float bev0 = be[0 * OUT_F + ocol], bev1 = be[1 * OUT_F + ocol];
            float bev2 = be[2 * OUT_F + ocol], bev3 = be[3 * OUT_F + ocol];
            #pragma unroll
            for (int j = 0; j < 8; ++j) {
                size_t row = blockM0 + rbase + j + 8 * lh;
                float v = ia * (baseacc[lt][j] + bbv + 4.0f * lacc[j])
                        + expacc[lt][j]
                        + wrow[0][j] * bev0 + wrow[1][j] * bev1
                        + wrow[2][j] * bev2 + wrow[3][j] * bev3;
                out[row * OUT_F + ocol] = v;
            }
        }
    }
}

// ---------------------------------------------------------------------------
extern "C" void kernel_launch(void* const* d_in, const int* in_sizes, int n_in,
                              void* d_out, int out_size, void* d_ws, size_t ws_size,
                              hipStream_t stream) {
    const float* x   = (const float*)d_in[0];
    const float* Wb  = (const float*)d_in[1];
    const float* bb  = (const float*)d_in[2];
    const float* A   = (const float*)d_in[3];
    const float* B   = (const float*)d_in[4];
    const float* ia3 = (const float*)d_in[5];
    const float* Wr  = (const float*)d_in[6];
    const float* br  = (const float*)d_in[7];
    const float* We  = (const float*)d_in[8];
    const float* be  = (const float*)d_in[9];
    float* out = (float*)d_out;
    unsigned short* wpk = (unsigned short*)d_ws;   // needs ~2.52 MB scratch

    int M = in_sizes[0] / IN_F;                    // 65536 tokens

    int nrep = (ENT1 + ENT2 + 255) / 256;
    repack_weights<<<nrep, 256, 0, stream>>>(Wb, We, Wr, A, wpk);

    poly_main<<<M / TM, 256, SM_TOTAL, stream>>>(x, bb, B, ia3, br, be, wpk, out);
}
